// ComENetEncoder_23089744183318
// MI455X (gfx1250) — matrix operand order repaired and checked by reading, not compiled, not run on hardware
//
#include <hip/hip_runtime.h>
#include <hip/hip_bf16.h>
#include <math.h>

#define CDIV(a,b) (((a)+(b)-1)/(b))
#define HIDC 128
#define NGR  128
#define CUTF 8.0f
#define PIF  3.14159265358979323846f

typedef __attribute__((ext_vector_type(16))) _Float16 v16h;
typedef __attribute__((ext_vector_type(8)))  _Float16 v8h;
typedef __attribute__((ext_vector_type(8)))  float    v8f;

union AV16 { v16h v; v8h h[2]; };

static __device__ __forceinline__ float siluf(float x){ return x / (1.0f + __expf(-x)); }
static __device__ __forceinline__ float sigmf(float x){ return 1.0f / (1.0f + __expf(-x)); }

static __device__ __forceinline__ v8f wmma32(v16h a, v16h b, v8f c){
  return __builtin_amdgcn_wmma_f32_16x16x32_f16(false, a, false, b, (short)0, c, false, false);
}

// A fragment (16x32 f16, M x K) from LDS row-major tile [16][ld].
// Lane l<16: M=l, K halves {kh*8..kh*8+7} and {16+kh*8..}, per ISA 7.12.2.
static __device__ __forceinline__ v16h frag_a(const _Float16* tile, int lane, int kbase, int ld){
  const int m = lane & 15, kh = (lane >> 4) & 1;
  const _Float16* rp = tile + m*ld + kbase + kh*8;
  AV16 a;
  a.h[0] = *(const v8h*)rp;        // K = kbase+kh*8 .. +7
  a.h[1] = *(const v8h*)(rp + 16); // K = kbase+16+kh*8 .. +7
  return a.v;
}

// B fragment (32x16, K x N) for Y = X @ W^T, W row-major [dout][din] f16 in LDS.
// B[k][n] = W[ntile*16+n][k]; lane holds column n, K halves {kh*8..kh*8+7} and {16+kh*8..}.
static __device__ __forceinline__ v16h frag_b(const _Float16* Ws, int lane, int ntile, int kbase, int din){
  const int n = lane & 15, kh = (lane >> 4) & 1;
  const _Float16* wr = Ws + (ntile*16 + n)*din + kbase + kh*8;
  AV16 b;
  b.h[0] = *(const v8h*)wr;
  b.h[1] = *(const v8h*)(wr + 16);
  return b.v;
}

// ---------------------------------------------------------------- elementwise

__global__ __launch_bounds__(256) void k_fill_u32(unsigned* p, unsigned v, int n){
  int i = blockIdx.x*blockDim.x + threadIdx.x;
  if (i < n) p[i] = v;
}

__global__ __launch_bounds__(256) void k_geom(const float* pos, const int* jl, const int* il,
                                              float* vecs, float* dist, int E){
  int e = blockIdx.x*blockDim.x + threadIdx.x;
  if (e >= E) return;
  int j = jl[e], i = il[e];
  float vx = pos[j*3+0] - pos[i*3+0];
  float vy = pos[j*3+1] - pos[i*3+1];
  float vz = pos[j*3+2] - pos[i*3+2];
  vecs[e*3+0] = vx; vecs[e*3+1] = vy; vecs[e*3+2] = vz;
  dist[e] = sqrtf(vx*vx + vy*vy + vz*vz);
}

__global__ __launch_bounds__(256) void k_minpass(const float* dist, const int* idx,
                                                 const unsigned* excl, unsigned* mb, int E){
  int e = blockIdx.x*blockDim.x + threadIdx.x;
  if (e >= E) return;
  int g = idx[e];
  float d = dist[e];
  if (excl != nullptr && excl[g] == (unsigned)e) d += CUTF;
  atomicMin(&mb[g], __float_as_uint(d));
}

__global__ __launch_bounds__(256) void k_argpass(const float* dist, const int* idx,
                                                 const unsigned* excl, const unsigned* mb,
                                                 unsigned* arg, int E){
  int e = blockIdx.x*blockDim.x + threadIdx.x;
  if (e >= E) return;
  int g = idx[e];
  float d = dist[e];
  if (excl != nullptr && excl[g] == (unsigned)e) d += CUTF;
  if (__float_as_uint(d) == mb[g]) atomicMin(&arg[g], (unsigned)e);
}

__global__ __launch_bounds__(256) void k_argfix(unsigned* arg, int n, int E){
  int i = blockIdx.x*blockDim.x + threadIdx.x;
  if (i < n && arg[i] >= (unsigned)E) arg[i] = 0u;
}

struct F3 { float x, y, z; };
static __device__ __forceinline__ F3 ldf3(const float* p){ F3 r; r.x=p[0]; r.y=p[1]; r.z=p[2]; return r; }
static __device__ __forceinline__ F3 negf3(F3 a){ F3 r; r.x=-a.x; r.y=-a.y; r.z=-a.z; return r; }
static __device__ __forceinline__ float dot3(F3 a, F3 b){ return a.x*b.x + a.y*b.y + a.z*b.z; }
static __device__ __forceinline__ F3 cross3(F3 a, F3 b){
  F3 r; r.x = a.y*b.z - a.z*b.y; r.y = a.z*b.x - a.x*b.z; r.z = a.x*b.y - a.y*b.x; return r;
}

__global__ __launch_bounds__(256) void k_feat(const float* vecs, const float* dist,
    const int* jl, const int* il,
    const unsigned* a0i, const unsigned* a1i, const unsigned* a0j, const unsigned* a1j,
    _Float16* emb1, _Float16* emb2, int E){
  int e = blockIdx.x*blockDim.x + threadIdx.x;
  if (e >= E) return;
  int i = il[e], j = jl[e];
  F3 pji = ldf3(vecs + 3*e);
  float d = dist[e];
  int e0i = (int)a0i[i], e1i = (int)a1i[i];
  int n0 = jl[e0i];
  F3 pin0 = ldf3(vecs + 3*e0i);
  F3 pin1 = ldf3(vecs + 3*e1i);
  int iref = (n0 == j) ? e1i : e0i;
  int e0j = (int)a0j[j], e1j = (int)a1j[j];
  int n0j = il[e0j];
  int jref = (n0j == i) ? e1j : e0j;
  F3 piref = ldf3(vecs + 3*iref);
  F3 pjref = ldf3(vecs + 3*jref);
  F3 npji = negf3(pji);
  // theta
  float ta = dot3(npji, pin0);
  F3 c0 = cross3(npji, pin0);
  float theta = atan2f(sqrtf(dot3(c0, c0)), ta);
  if (theta < 0.f) theta += PIF;
  // phi
  F3 p1 = cross3(npji, pin0), p2 = cross3(npji, pin1);
  float phi = atan2f(dot3(cross3(p1, p2), pji) / d, dot3(p1, p2));
  if (phi < 0.f) phi += PIF;
  // tau
  F3 q1 = cross3(pji, pjref), q2 = cross3(pji, piref);
  float tau = atan2f(dot3(cross3(q1, q2), pji) / d, dot3(q1, q2));
  if (tau < 0.f) tau += PIF;
  float s = sqrtf(2.0f / CUTF);
  float ct = cosf(theta), cp = cosf(phi), cu = cosf(tau);
  #pragma unroll
  for (int r = 0; r < 3; ++r){
    float rb = s * sinf((float)(r+1) * PIF * d / CUTF) / d;
    emb1[e*16 + r*4 + 0] = (_Float16)rb;
    emb1[e*16 + r*4 + 1] = (_Float16)(rb*cp);
    emb1[e*16 + r*4 + 2] = (_Float16)(rb*ct);
    emb1[e*16 + r*4 + 3] = (_Float16)(rb*ct*cp);
    emb2[e*8 + r*2 + 0]  = (_Float16)rb;
    emb2[e*8 + r*2 + 1]  = (_Float16)(rb*cu);
  }
}

__global__ __launch_bounds__(256) void k_embed(const int* z, const float* emb, float* x, int N){
  int idx = blockIdx.x*blockDim.x + threadIdx.x;
  if (idx >= N*HIDC) return;
  int n = idx >> 7, c = idx & 127;
  x[idx] = siluf(emb[z[n]*HIDC + c]);
}

// ------------------------------------------------------------- graph norm

__global__ __launch_bounds__(256) void k_gn1(const float* h, const int* batch,
                                             float* sum, float* cnt, int N){
  int idx = blockIdx.x*blockDim.x + threadIdx.x;
  if (idx >= N*HIDC) return;
  int n = idx >> 7, c = idx & 127;
  int g = batch[n];
  atomicAdd(&sum[g*HIDC + c], h[idx]);
  if (c == 0) atomicAdd(&cnt[g], 1.0f);
}

__global__ __launch_bounds__(256) void k_gn2(const float* h, const int* batch,
                                             const float* sum, const float* cnt,
                                             const float* mscale, float* tmp, float* sq, int N){
  int idx = blockIdx.x*blockDim.x + threadIdx.x;
  if (idx >= N*HIDC) return;
  int n = idx >> 7, c = idx & 127;
  int g = batch[n];
  float ct = fmaxf(cnt[g], 1.0f);
  float mean = sum[g*HIDC + c] / ct;
  float o = h[idx] - mean * mscale[c];
  tmp[idx] = o;
  atomicAdd(&sq[g*HIDC + c], o*o);
}

__global__ __launch_bounds__(256) void k_gn3(const float* tmp, const int* batch,
                                             const float* sq, const float* cnt,
                                             const float* wgt, const float* bia, float* h, int N){
  int idx = blockIdx.x*blockDim.x + threadIdx.x;
  if (idx >= N*HIDC) return;
  int n = idx >> 7, c = idx & 127;
  int g = batch[n];
  float var = sq[g*HIDC + c] / fmaxf(cnt[g], 1.0f);
  h[idx] = tmp[idx] * rsqrtf(var + 1e-5f) * wgt[c] + bia[c];
}

// ------------------------------------------------------------- WMMA kernels

// Small MLP: out = silu(W2 @ silu(W1 @ emb + b1) + b2), per-edge, f16 out.
template<int DINR, int DSTR>
__global__ __launch_bounds__(128) void k_mlp_small(const _Float16* embh,
    const float* W1, const float* b1, const float* W2, const float* b2,
    _Float16* outh, int E){
  __shared__ __attribute__((aligned(16))) _Float16 W1s[64*32];
  __shared__ __attribute__((aligned(16))) _Float16 W2s[128*64];
  __shared__ __attribute__((aligned(16))) _Float16 As[4][16*32];
  __shared__ __attribute__((aligned(16))) _Float16 Ms[4][16*64];
  const int tid = threadIdx.x, lane = tid & 31, wid = tid >> 5;
  for (int idx = tid; idx < 64*32; idx += 128){
    int r = idx >> 5, k = idx & 31;
    W1s[idx] = (_Float16)((k < DINR) ? W1[r*DINR + k] : 0.0f);
  }
  for (int idx = tid; idx < 128*64; idx += 128) W2s[idx] = (_Float16)W2[idx];
  __syncthreads();
  const int col0 = lane & 15, kh = (lane >> 4) & 1;
  const int ntile = CDIV(E, 16);
  for (int t = blockIdx.x*4 + wid; t < ntile; t += gridDim.x*4){
    int eb = t*16;
    for (int idx = lane; idx < 16*32; idx += 32){
      int r = idx >> 5, k = idx & 31;
      int e = eb + r;
      _Float16 v = (_Float16)0.0f;
      if (e < E && k < DINR) v = embh[e*DSTR + k];
      As[wid][idx] = v;
    }
    v8f acc1[4] = {};
    {
      v16h a = frag_a(As[wid], lane, 0, 32);
      #pragma unroll
      for (int n = 0; n < 4; ++n) acc1[n] = wmma32(a, frag_b(W1s, lane, n, 0, 32), acc1[n]);
    }
    #pragma unroll
    for (int n = 0; n < 4; ++n)
      #pragma unroll
      for (int r = 0; r < 8; ++r){
        int m = r + 8*kh, c = n*16 + col0;
        Ms[wid][m*64 + c] = (_Float16)siluf(acc1[n][r] + b1[c]);
      }
    v8f acc2[8] = {};
    #pragma unroll
    for (int kb = 0; kb < 64; kb += 32){
      v16h a = frag_a(Ms[wid], lane, kb, 64);
      #pragma unroll
      for (int n = 0; n < 8; ++n) acc2[n] = wmma32(a, frag_b(W2s, lane, n, kb, 64), acc2[n]);
    }
    #pragma unroll
    for (int n = 0; n < 8; ++n)
      #pragma unroll
      for (int r = 0; r < 8; ++r){
        int m = r + 8*kh, c = n*16 + col0;
        int e = eb + m;
        if (e < E) outh[e*HIDC + c] = (_Float16)siluf(acc2[n][r] + b2[c]);
      }
  }
}

// Generic node GEMM: Y = act(X @ W^T + b) + add;  X = [X1 | X2], DOUT = 128.
template<int DIN1, int DIN2, bool ACT>
__global__ __launch_bounds__(128) void k_node_gemm(const float* X1, const float* X2,
    const float* W, const float* bias, const float* add, float* Y, int rows){
  constexpr int DIN = DIN1 + DIN2;
  constexpr int DIN2X = (DIN2 > 0) ? DIN2 : 1;
  __shared__ __attribute__((aligned(16))) _Float16 Ws[128*DIN];
  __shared__ __attribute__((aligned(16))) _Float16 As[4][16*DIN];
  const int tid = threadIdx.x, lane = tid & 31, wid = tid >> 5;
  for (int idx = tid; idx < 128*DIN; idx += 128) Ws[idx] = (_Float16)W[idx];
  __syncthreads();
  const int col0 = lane & 15, kh = (lane >> 4) & 1;
  const int ntile = CDIV(rows, 16);
  for (int t = blockIdx.x*4 + wid; t < ntile; t += gridDim.x*4){
    int rb = t*16;
    for (int idx = lane; idx < 16*DIN; idx += 32){
      int r = idx / DIN, c = idx % DIN;
      int row = rb + r;
      float v = 0.0f;
      if (row < rows) v = (c < DIN1) ? X1[row*DIN1 + c] : X2[row*DIN2X + (c - DIN1)];
      As[wid][idx] = (_Float16)v;
    }
    v8f acc[8] = {};
    #pragma unroll
    for (int kb = 0; kb < DIN; kb += 32){
      v16h a = frag_a(As[wid], lane, kb, DIN);
      #pragma unroll
      for (int n = 0; n < 8; ++n) acc[n] = wmma32(a, frag_b(Ws, lane, n, kb, DIN), acc[n]);
    }
    #pragma unroll
    for (int n = 0; n < 8; ++n)
      #pragma unroll
      for (int r = 0; r < 8; ++r){
        int m = r + 8*kh, c = n*16 + col0;
        int row = rb + m;
        if (row < rows){
          float v = acc[n][r];
          if (bias != nullptr) v += bias[c];
          if (ACT) v = siluf(v);
          if (add != nullptr) v += add[row*HIDC + c];
          Y[row*HIDC + c] = v;
        }
      }
  }
}

// Fused edge conv: ew = el2(silu(el1([ebond|fgeo]))); m = ew*x[src];
// m *= sigmoid(attn.m); atomic agg[dst] += m.
__global__ __launch_bounds__(128) void k_conv_edge(const float* x,
    const int* src, const int* dst, const float* ebond, const _Float16* fgeo,
    const float* Wel1, const float* bel1, const float* Wel2, const float* bel2,
    const float* Wattn, const float* battn, float* agg, int E){
  __shared__ __attribute__((aligned(16))) _Float16 W1s[128*256];
  __shared__ __attribute__((aligned(16))) _Float16 W2s[128*128];
  __shared__ float wats[128];
  __shared__ __attribute__((aligned(16))) unsigned char scratch[4][16*256*2]; // 8KB/wave, overlaid A/mid/ew
  const int tid = threadIdx.x, lane = tid & 31, wid = tid >> 5;
  for (int idx = tid; idx < 128*256; idx += 128) W1s[idx] = (_Float16)Wel1[idx];
  for (int idx = tid; idx < 128*128; idx += 128) W2s[idx] = (_Float16)Wel2[idx];
  if (tid < 128) wats[tid] = Wattn[tid];
  __syncthreads();
  const float b0 = battn[0];
  _Float16* As  = (_Float16*)scratch[wid];
  _Float16* Ms  = (_Float16*)scratch[wid];
  float*    ews = (float*)scratch[wid];
  const int col0 = lane & 15, kh = (lane >> 4) & 1;
  const int ntile = CDIV(E, 16);
  for (int t = blockIdx.x*4 + wid; t < ntile; t += gridDim.x*4){
    int eb = t*16;
    // stage A [16][256] f16: [ebond | fgeo]
    for (int idx = lane; idx < 16*256; idx += 32){
      int r = idx >> 8, c = idx & 255;
      int e = eb + r;
      _Float16 v = (_Float16)0.0f;
      if (e < E) v = (c < 128) ? (_Float16)ebond[e*HIDC + c] : fgeo[e*HIDC + (c - 128)];
      As[idx] = v;
    }
    // GEMM1: K=256
    v8f acc[8] = {};
    #pragma unroll
    for (int kb = 0; kb < 256; kb += 32){
      v16h a = frag_a(As, lane, kb, 256);
      #pragma unroll
      for (int n = 0; n < 8; ++n) acc[n] = wmma32(a, frag_b(W1s, lane, n, kb, 256), acc[n]);
    }
    #pragma unroll
    for (int n = 0; n < 8; ++n)
      #pragma unroll
      for (int r = 0; r < 8; ++r){
        int m = r + 8*kh, c = n*16 + col0;
        Ms[m*128 + c] = (_Float16)siluf(acc[n][r] + bel1[c]);
      }
    // GEMM2: K=128
    v8f acc2[8] = {};
    #pragma unroll
    for (int kb = 0; kb < 128; kb += 32){
      v16h a = frag_a(Ms, lane, kb, 128);
      #pragma unroll
      for (int n = 0; n < 8; ++n) acc2[n] = wmma32(a, frag_b(W2s, lane, n, kb, 128), acc2[n]);
    }
    #pragma unroll
    for (int n = 0; n < 8; ++n)
      #pragma unroll
      for (int r = 0; r < 8; ++r){
        int m = r + 8*kh, c = n*16 + col0;
        ews[m*128 + c] = acc2[n][r] + bel2[c];
      }
    // gather / gate / scatter
    for (int r = 0; r < 16; ++r){
      int e = eb + r;
      if (e >= E) break;
      int s = src[e], dd = dst[e];
      int c0 = lane * 4;
      float4 xv = *(const float4*)(x + s*HIDC + c0);
      float m0 = ews[r*128 + c0 + 0] * xv.x;
      float m1 = ews[r*128 + c0 + 1] * xv.y;
      float m2 = ews[r*128 + c0 + 2] * xv.z;
      float m3 = ews[r*128 + c0 + 3] * xv.w;
      float part = m0*wats[c0] + m1*wats[c0+1] + m2*wats[c0+2] + m3*wats[c0+3];
      #pragma unroll
      for (int o = 16; o > 0; o >>= 1) part += __shfl_xor(part, o, 32);
      float g = sigmf(part + b0);
      float* ag = agg + dd*HIDC + c0;
      atomicAdd(ag + 0, m0*g);
      atomicAdd(ag + 1, m1*g);
      atomicAdd(ag + 2, m2*g);
      atomicAdd(ag + 3, m3*g);
    }
  }
}

// ---------------------------------------------------------------- host side

struct ConvP { const float *We1,*be1,*We2,*be2,*Wat,*bat,*Wrel,*brel,*Wroot; };
struct BlockP {
  const float *Wlin,*blin;
  ConvP c[2];
  const float *Wl1,*bl1,*Wl2,*bl2,*Wcat,*bcat;
  const float *Wr0,*br0,*Wr1,*br1;
  const float *gw,*gb,*gms;
  const float *Wf,*bf;
};

extern "C" void kernel_launch(void* const* d_in, const int* in_sizes, int n_in,
                              void* d_out, int out_size, void* d_ws, size_t ws_size,
                              hipStream_t stream){
  (void)n_in; (void)out_size; (void)ws_size;
  const int N = in_sizes[0];
  const int E = in_sizes[3] / HIDC;

  const int*   z     = (const int*)d_in[0];
  const float* pos   = (const float*)d_in[1];
  const int*   jl    = (const int*)d_in[2];
  const int*   il    = jl + E;
  const float* ebond = (const float*)d_in[3];
  const int*   batch = (const int*)d_in[4];

  int pc = 5;
  auto P = [&]() { return (const float*)d_in[pc++]; };
  const float* embW = P();
  const float *f1W1 = P(), *f1b1 = P(), *f1W2 = P(), *f1b2 = P();
  const float *f2W1 = P(), *f2b1 = P(), *f2W2 = P(), *f2b2 = P();
  BlockP B[2];
  for (int b = 0; b < 2; ++b){
    B[b].Wlin = P(); B[b].blin = P();
    for (int c = 0; c < 2; ++c){
      ConvP cp;
      cp.We1 = P(); cp.be1 = P(); cp.We2 = P(); cp.be2 = P();
      cp.Wat = P(); cp.bat = P(); cp.Wrel = P(); cp.brel = P(); cp.Wroot = P();
      B[b].c[c] = cp;
    }
    B[b].Wl1 = P(); B[b].bl1 = P(); B[b].Wl2 = P(); B[b].bl2 = P();
    B[b].Wcat = P(); B[b].bcat = P();
    B[b].Wr0 = P(); B[b].br0 = P(); B[b].Wr1 = P(); B[b].br1 = P();
    B[b].gw = P(); B[b].gb = P(); B[b].gms = P();
    B[b].Wf = P(); B[b].bf = P();
  }

  // workspace carve
  char* wp = (char*)d_ws;
  auto A = [&](size_t bytes){ void* r = (void*)wp; wp += (bytes + 255) & ~(size_t)255; return r; };
  float*    vecs  = (float*)A((size_t)E*3*4);
  float*    dist  = (float*)A((size_t)E*4);
  unsigned* mb0i  = (unsigned*)A((size_t)N*4);
  unsigned* a0i   = (unsigned*)A((size_t)N*4);
  unsigned* mb1i  = (unsigned*)A((size_t)N*4);
  unsigned* a1i   = (unsigned*)A((size_t)N*4);
  unsigned* mb0j  = (unsigned*)A((size_t)N*4);
  unsigned* a0j   = (unsigned*)A((size_t)N*4);
  unsigned* mb1j  = (unsigned*)A((size_t)N*4);
  unsigned* a1j   = (unsigned*)A((size_t)N*4);
  _Float16* emb1h = (_Float16*)A((size_t)E*16*2);
  _Float16* emb2h = (_Float16*)A((size_t)E*8*2);
  _Float16* f1h   = (_Float16*)A((size_t)E*HIDC*2);
  _Float16* f2h   = (_Float16*)A((size_t)E*HIDC*2);
  float*    agg   = (float*)A((size_t)N*HIDC*4);
  float*    xbuf  = (float*)A((size_t)N*HIDC*4);
  float*    xl    = (float*)A((size_t)N*HIDC*4);
  float*    cbuf  = (float*)A((size_t)N*HIDC*4);
  float*    h1    = (float*)A((size_t)N*HIDC*4);
  float*    h2    = (float*)A((size_t)N*HIDC*4);
  float*    hbuf  = (float*)A((size_t)N*HIDC*4);
  float*    gsum  = (float*)A((size_t)NGR*HIDC*4);
  float*    gsq   = (float*)A((size_t)NGR*HIDC*4);
  float*    gcnt  = (float*)A((size_t)NGR*4);

  const int TB = 256;
  const int egrid = CDIV(E, TB);
  const int ngrid_elt = CDIV(N*HIDC, TB);
  const int ngrid = CDIV(CDIV(N, 16), 4);   // node-GEMM tiles / 4 waves
  const int mgrid = 640;                    // edge-tile WMMA kernels

  k_geom<<<egrid, TB, 0, stream>>>(pos, jl, il, vecs, dist, E);

  auto seg_argmin = [&](const int* idx, unsigned* mb, unsigned* arg, const unsigned* excl){
    k_fill_u32<<<CDIV(N, TB), TB, 0, stream>>>(mb, 0x7F800000u, N);
    k_fill_u32<<<CDIV(N, TB), TB, 0, stream>>>(arg, 0xFFFFFFFFu, N);
    k_minpass<<<egrid, TB, 0, stream>>>(dist, idx, excl, mb, E);
    k_argpass<<<egrid, TB, 0, stream>>>(dist, idx, excl, mb, arg, E);
    k_argfix<<<CDIV(N, TB), TB, 0, stream>>>(arg, N, E);
  };
  seg_argmin(il, mb0i, a0i, nullptr);
  seg_argmin(il, mb1i, a1i, a0i);
  seg_argmin(jl, mb0j, a0j, nullptr);
  seg_argmin(jl, mb1j, a1j, a0j);

  k_feat<<<egrid, TB, 0, stream>>>(vecs, dist, jl, il, a0i, a1i, a0j, a1j, emb1h, emb2h, E);
  k_mlp_small<12,16><<<mgrid, 128, 0, stream>>>(emb1h, f1W1, f1b1, f1W2, f1b2, f1h, E);
  k_mlp_small< 6, 8><<<mgrid, 128, 0, stream>>>(emb2h, f2W1, f2b1, f2W2, f2b2, f2h, E);
  k_embed<<<ngrid_elt, TB, 0, stream>>>(z, embW, xbuf, N);

  float* xin = xbuf;
  for (int b = 0; b < 2; ++b){
    const BlockP& bp = B[b];
    // x = silu(lin(x))
    k_node_gemm<128,0,true><<<ngrid, 128, 0, stream>>>(xin, nullptr, bp.Wlin, bp.blin, nullptr, xl, N);
    for (int c = 0; c < 2; ++c){
      const ConvP& cp = bp.c[c];
      k_fill_u32<<<ngrid_elt, TB, 0, stream>>>((unsigned*)agg, 0u, N*HIDC);
      k_conv_edge<<<mgrid, 128, 0, stream>>>(xl, jl, il, ebond, (c == 0) ? f1h : f2h,
          cp.We1, cp.be1, cp.We2, cp.be2, cp.Wat, cp.bat, agg, E);
      // conv_out = lin_rel(agg) + lin_root(xl)
      k_node_gemm<128,0,false><<<ngrid, 128, 0, stream>>>(agg, nullptr, cp.Wrel, cp.brel, nullptr, cbuf, N);
      k_node_gemm<128,0,false><<<ngrid, 128, 0, stream>>>(xl, nullptr, cp.Wroot, nullptr, cbuf, cbuf, N);
      // h{1,2} = silu(lin{1,2}(conv_out))
      const float* Wl = (c == 0) ? bp.Wl1 : bp.Wl2;
      const float* bl = (c == 0) ? bp.bl1 : bp.bl2;
      k_node_gemm<128,0,true><<<ngrid, 128, 0, stream>>>(cbuf, nullptr, Wl, bl, nullptr,
                                                         (c == 0) ? h1 : h2, N);
    }
    // h = lin_cat([h1|h2]) + x
    k_node_gemm<128,128,false><<<ngrid, 128, 0, stream>>>(h1, h2, bp.Wcat, bp.bcat, xl, hbuf, N);
    // residual linears: h = silu(lin(h)) + h
    k_node_gemm<128,0,true><<<ngrid, 128, 0, stream>>>(hbuf, nullptr, bp.Wr0, bp.br0, hbuf, hbuf, N);
    k_node_gemm<128,0,true><<<ngrid, 128, 0, stream>>>(hbuf, nullptr, bp.Wr1, bp.br1, hbuf, hbuf, N);
    // graph norm
    k_fill_u32<<<CDIV(NGR*HIDC, TB), TB, 0, stream>>>((unsigned*)gsum, 0u, NGR*HIDC);
    k_fill_u32<<<CDIV(NGR*HIDC, TB), TB, 0, stream>>>((unsigned*)gsq, 0u, NGR*HIDC);
    k_fill_u32<<<1, TB, 0, stream>>>((unsigned*)gcnt, 0u, NGR);
    k_gn1<<<ngrid_elt, TB, 0, stream>>>(hbuf, batch, gsum, gcnt, N);
    k_gn2<<<ngrid_elt, TB, 0, stream>>>(hbuf, batch, gsum, gcnt, bp.gms, cbuf, gsq, N);
    k_gn3<<<ngrid_elt, TB, 0, stream>>>(cbuf, batch, gsq, gcnt, bp.gw, bp.gb, hbuf, N);
    // x = final(h)
    float* xo = (b == 1) ? (float*)d_out : xbuf;
    k_node_gemm<128,0,false><<<ngrid, 128, 0, stream>>>(hbuf, nullptr, bp.Wf, bp.bf, nullptr, xo, N);
    xin = xbuf;
  }
}
